// Transformer_67430986547344
// MI455X (gfx1250) — matrix operand order repaired and checked by reading, not compiled, not run on hardware
//
#include <hip/hip_runtime.h>

#define C_DIM 128
#define NHD   8
#define HN    64
#define S2D   128   // 2*HN
#define W_FULL 128
#define W_LOW  64
#define LYR   6

typedef __attribute__((ext_vector_type(16))) __bf16 v16bf;
typedef __attribute__((ext_vector_type(8)))  float  v8f;
typedef int v4i __attribute__((vector_size(16)));

// ---------------- helpers ----------------
__device__ __forceinline__ unsigned short f2bfu(float x){
  union { float f; unsigned u; } c; c.f = x;
  unsigned r = c.u + 0x7FFFu + ((c.u >> 16) & 1u);
  return (unsigned short)(r >> 16);
}
__device__ __forceinline__ __bf16 f2bf(float x){
  union { unsigned short s; __bf16 b; } o; o.s = f2bfu(x); return o.b;
}
__device__ __forceinline__ __bf16 bf_bits(unsigned short s){
  union { unsigned short s; __bf16 b; } o; o.s = s; return o.b;
}
__device__ __forceinline__ float wsum(float v){
#pragma unroll
  for (int m = 16; m >= 1; m >>= 1) v += __shfl_xor(v, m, 32);
  return v;
}
__device__ __forceinline__ float wmaxr(float v){
#pragma unroll
  for (int m = 16; m >= 1; m >>= 1) v = fmaxf(v, __shfl_xor(v, m, 32));
  return v;
}
#define WMMA_BF16(a, b, c) \
  __builtin_amdgcn_wmma_f32_16x16x32_bf16(false, (a), false, (b), (short)0, (c), false, false)

// ---- CDNA5 async global->LDS copy (ASYNCcnt) with safe fallback ----
#if __has_builtin(__builtin_amdgcn_global_load_async_to_lds_b128) && \
    __has_builtin(__builtin_amdgcn_s_wait_asynccnt)
#define HAS_ASYNC 1
__device__ __forceinline__ void async_cp16(const float* g, float* l){
  __builtin_amdgcn_global_load_async_to_lds_b128(
      (__attribute__((address_space(1))) v4i*)g,
      (__attribute__((address_space(3))) v4i*)l, 0, 0);
}
__device__ __forceinline__ void async_wait(){
  __builtin_amdgcn_s_wait_asynccnt(0);
}
#else
#define HAS_ASYNC 0
__device__ __forceinline__ void async_cp16(const float* g, float* l){
  l[0] = g[0]; l[1] = g[1]; l[2] = g[2]; l[3] = g[3];
}
__device__ __forceinline__ void async_wait(){}
#endif

// ---------------- elementwise kernels ----------------
// feat[w, s, c]; s<Hd -> left[0,c,s,w], else right[0,c,s-Hd,w]
__global__ void k_flat(const float* __restrict__ left, const float* __restrict__ right,
                       float* __restrict__ out, int Wd, int Hd){
  int idx = blockIdx.x * blockDim.x + threadIdx.x;
  int total = Wd * 2 * Hd * C_DIM;
  if (idx >= total) return;
  int c = idx % C_DIM;
  int s = (idx / C_DIM) % (2 * Hd);
  int w = idx / (C_DIM * 2 * Hd);
  const float* src = (s < Hd) ? left : right;
  int h = (s < Hd) ? s : s - Hd;
  out[idx] = src[(c * Hd + h) * Wd + w];
}

// LayerNorm over C=128; one wave per row (l,b). out contiguous [L,B,C].
__global__ void k_ln(const float* __restrict__ in, long inOff, long ldiL, int Brow,
                     const float* __restrict__ g, const float* __restrict__ bta,
                     float* __restrict__ out, int L){
  int wv = (blockIdx.x * blockDim.x + threadIdx.x) >> 5;
  int lane = threadIdx.x & 31;
  if (wv >= L * Brow) return;
  int l = wv / Brow, bb = wv % Brow;
  const float* src = in + inOff + (long)l * ldiL + (long)bb * C_DIM;
  float x[4];
#pragma unroll
  for (int i = 0; i < 4; i++) x[i] = src[lane * 4 + i];
  float mean = wsum(x[0] + x[1] + x[2] + x[3]) * (1.f / C_DIM);
  float vs = 0.f;
#pragma unroll
  for (int i = 0; i < 4; i++){ float d = x[i] - mean; vs += d * d; }
  float inv = rsqrtf(wsum(vs) * (1.f / C_DIM) + 1e-5f);
  float* dst = out + (long)wv * C_DIM;
#pragma unroll
  for (int i = 0; i < 4; i++){
    int c = lane * 4 + i;
    dst[c] = (x[i] - mean) * inv * g[c] + bta[c];
  }
}

// y-permute: y[j, side*Wd+i, c] <-> x[i, side*HN+j, c]   (HN=64, S=128)
__global__ void k_permy(const float* __restrict__ in, float* __restrict__ out, int Wd, int fwd){
  int idx = blockIdx.x * blockDim.x + threadIdx.x;
  int total = Wd * S2D * C_DIM;
  if (idx >= total) return;
  int c = idx % C_DIM;
  int t = idx / C_DIM;
  int j = t % HN;
  int side = (t / HN) & 1;
  int i = t / S2D;
  long xi = ((long)i * S2D + side * HN + j) * C_DIM + c;
  long yi = ((long)j * (2 * Wd) + side * Wd + i) * C_DIM + c;
  if (fwd) out[yi] = in[xi]; else out[xi] = in[yi];
}

// strided (l,b,C) copy
__global__ void k_copy(const float* __restrict__ in, long inOff, long ldiL,
                       float* __restrict__ out, long outOff, long ldoL, int Brow, int L){
  int idx = blockIdx.x * blockDim.x + threadIdx.x;
  int total = L * Brow * C_DIM;
  if (idx >= total) return;
  int c = idx % C_DIM;
  int bb = (idx / C_DIM) % Brow;
  int l = idx / (C_DIM * Brow);
  out[outOff + (long)l * ldoL + (long)bb * C_DIM + c] =
      in[inOff + (long)l * ldiL + (long)bb * C_DIM + c];
}

// conv_in[w,s,ch]: ch<128 -> linear-interp(fn, dim W1->W), else feat[w,s,ch-128]
__global__ void k_interpcat(const float* __restrict__ fn, const float* __restrict__ feat,
                            float* __restrict__ out){
  int idx = blockIdx.x * blockDim.x + threadIdx.x;
  int total = W_FULL * S2D * 256;
  if (idx >= total) return;
  int ch = idx % 256;
  int s = (idx / 256) % S2D;
  int w = idx / (256 * S2D);
  float v;
  if (ch < 128){
    float coord = (w + 0.5f) * ((float)W_LOW / (float)W_FULL) - 0.5f;
    coord = fminf(fmaxf(coord, 0.f), (float)(W_LOW - 1));
    int lo = (int)floorf(coord);
    int hi = lo + 1 < W_LOW - 1 ? lo + 1 : W_LOW - 1;
    float fr = coord - (float)lo;
    v = fn[((long)lo * S2D + s) * C_DIM + ch] * (1.f - fr) +
        fn[((long)hi * S2D + s) * C_DIM + ch] * fr;
  } else {
    v = feat[((long)w * S2D + s) * C_DIM + (ch - 128)];
  }
  out[idx] = v;
}

// ---------------- generic WMMA GEMM: out = A[M,K] * W[N,K]^T + bias ----------------
// A/B tiles staged to LDS via async b128 copies; per-wave 16x16 output tile.
// out token m -> outOff + (m/outB)*ldoL + (m%outB)*N + n ; acc: accumulate
__global__ void k_gemm(const float* __restrict__ A, int M, int K,
                       const float* __restrict__ Wt, const float* __restrict__ bias, int N,
                       float* __restrict__ out, long outOff, long ldoL, int outB, int acc){
  __shared__ float lA[4][16 * 32];
  __shared__ float lB[4][16 * 32];
  int wid = threadIdx.x >> 5;
  int lane = threadIdx.x & 31;
  int wv = blockIdx.x * 4 + wid;
  int mt = (M + 15) / 16, nt = N / 16;
  if (wv >= mt * nt) return;
  int m0 = (wv / nt) * 16, n0 = (wv % nt) * 16;
  int half = lane >> 4, ln = lane & 15;
  v8f zero = {};
  union { v8f v; float e[8]; } cc; cc.v = zero;
  for (int k0 = 0; k0 < K; k0 += 32){
    // stage 16x32 A tile and 16x32 B tile, 16B per lane per copy
    for (int idx = lane; idx < 16 * 8; idx += 32){
      int row = idx >> 3, c4 = (idx & 7) * 4;
      int ar = m0 + row; if (ar >= M) ar = M - 1;
      async_cp16(&A[(long)ar * K + k0 + c4], &lA[wid][row * 32 + c4]);
      async_cp16(&Wt[(long)(n0 + row) * K + k0 + c4], &lB[wid][row * 32 + c4]);
    }
    async_wait();
    union { v16bf v; __bf16 e[16]; } a, b;
#pragma unroll
    for (int i = 0; i < 16; i++){
      int ka = i + 8 * ((i >> 3) + half);
      a.e[i] = f2bf(lA[wid][ln * 32 + ka]);
      int kb = i + 8 * ((i >> 3) + half);
      b.e[i] = f2bf(lB[wid][ln * 32 + kb]);
    }
    cc.v = WMMA_BF16(a.v, b.v, cc.v);
  }
#pragma unroll
  for (int r = 0; r < 8; r++){
    int m = m0 + r + 8 * half;
    if (m >= M) continue;
    int n = n0 + ln;
    float v = cc.e[r] + (bias ? bias[n] : 0.f);
    long o = outOff + (long)(m / outB) * ldoL + (long)(m % outB) * N + n;
    if (acc) out[o] += v; else out[o] = v;
  }
}

// ---------------- fused relative attention ----------------
// qkvQ/qkvKV: [L,B,384] (q|k|v, unscaled). pe: [2L-1,256] (q-proj | k-proj, unscaled).
// S[w,v] = scale*(q·k + q·PEk[d] + PEq[d]·k) (+ causal -inf), d = flip? L-1+w-v : L-1-w+v
// vo: [L,B,C]. rawOut (optional): [B,L,L] head-summed pre-softmax scores.
__global__ void __launch_bounds__(32)
k_attn(const float* __restrict__ qkvQ, const float* __restrict__ qkvKV,
       const float* __restrict__ pe, float* __restrict__ vo, float* __restrict__ rawOut,
       int L, int B, int flip, int causal, float scale){
  __shared__ float lQ[16 * 16];
  __shared__ float lK[128 * 16];
  __shared__ float lV[128 * 16];
  __shared__ float lPq[144 * 16];
  __shared__ float lPk[144 * 16];
  __shared__ float lS[16 * 128];
  __shared__ float lRaw[16 * 128];
  __shared__ unsigned short lP[16 * 128];

  int w0 = blockIdx.x * 16;
  int nb = blockIdx.y;
  int lane = threadIdx.x;
  int half = lane >> 4, ln = lane & 15;
  int n1 = L - 1;
  int d0 = flip ? (n1 + w0 - (L - 1)) : (n1 - w0 - 15);
  int band = L + 15;
  v8f zero = {};

  for (int e = 0; e < NHD; e++){
    // ---- async-stage Q tile, K/V columns, PE bands into LDS ----
    for (int idx = lane; idx < 16 * 4; idx += 32){
      int row = idx >> 2, c4 = (idx & 3) * 4;
      async_cp16(&qkvQ[((long)(w0 + row) * B + nb) * 384 + e * 16 + c4], &lQ[row * 16 + c4]);
    }
    for (int idx = lane; idx < L * 4; idx += 32){
      int row = idx >> 2, c4 = (idx & 3) * 4;
      long base = ((long)row * B + nb) * 384 + e * 16 + c4;
      async_cp16(&qkvKV[base + 128], &lK[row * 16 + c4]);
      async_cp16(&qkvKV[base + 256], &lV[row * 16 + c4]);
    }
    for (int idx = lane; idx < band * 4; idx += 32){
      int row = idx >> 2, c4 = (idx & 3) * 4;
      int d = d0 + row;                 // always within [0, 2L-2]
      async_cp16(&pe[(long)d * 256 + e * 16 + c4], &lPq[row * 16 + c4]);
      async_cp16(&pe[(long)d * 256 + 128 + e * 16 + c4], &lPk[row * 16 + c4]);
    }
    async_wait();
    __syncthreads();

    // S1 = (scale*Q) * K^T  (K-dim 16 padded to 32)
    union { v16bf v; __bf16 e[16]; } aq;
#pragma unroll
    for (int i = 0; i < 16; i++){
      int k = i + 8 * ((i >> 3) + half);
      aq.e[i] = (k < 16) ? f2bf(scale * lQ[ln * 16 + k]) : f2bf(0.f);
    }
    for (int v0 = 0; v0 < L; v0 += 16){
      union { v16bf v; __bf16 e[16]; } bk;
#pragma unroll
      for (int i = 0; i < 16; i++){
        int k = i + 8 * ((i >> 3) + half);
        bk.e[i] = (k < 16) ? f2bf(lK[(v0 + ln) * 16 + k]) : f2bf(0.f);
      }
      union { v8f v; float e[8]; } cc; cc.v = zero;
      cc.v = WMMA_BF16(aq.v, bk.v, cc.v);
#pragma unroll
      for (int r = 0; r < 8; r++) lS[(r + 8 * half) * 128 + v0 + ln] = cc.e[r];
    }
    __syncthreads();

    // relative terms + mask + raw accumulation (VALU, scale at use)
    for (int idx = lane; idx < 16 * L; idx += 32){
      int wl = idx / L, v = idx % L;
      float acc = lS[wl * 128 + v];
      int j = flip ? (wl + L - 1 - v) : (15 - wl + v);
      float a1 = 0.f, a2 = 0.f;
#pragma unroll
      for (int c = 0; c < 16; c++){
        a1 += lQ[wl * 16 + c] * lPk[j * 16 + c];
        a2 += lK[v * 16 + c] * lPq[j * 16 + c];
      }
      acc += scale * (a1 + a2);
      if (causal && v > w0 + wl) acc = -__builtin_inff();
      lS[wl * 128 + v] = acc;
      if (rawOut){
        float rr = (e == 0) ? 0.f : lRaw[wl * 128 + v];
        lRaw[wl * 128 + v] = rr + acc;
      }
    }
    __syncthreads();

    // softmax rows
    for (int r = 0; r < 16; r++){
      float mx = -__builtin_inff();
      for (int v = lane; v < L; v += 32) mx = fmaxf(mx, lS[r * 128 + v]);
      mx = wmaxr(mx);
      float sm = 0.f;
      for (int v = lane; v < L; v += 32) sm += __expf(lS[r * 128 + v] - mx);
      sm = wsum(sm);
      float inv = 1.f / sm;
      for (int v = lane; v < L; v += 32)
        lP[r * 128 + v] = f2bfu(__expf(lS[r * 128 + v] - mx) * inv);
    }
    __syncthreads();

    // O = P * V
    union { v8f v; float e[8]; } oo; oo.v = zero;
    for (int v0 = 0; v0 < L; v0 += 32){
      union { v16bf v; __bf16 e[16]; } pa, vb;
#pragma unroll
      for (int i = 0; i < 16; i++){
        int ka = i + 8 * ((i >> 3) + half);
        pa.e[i] = bf_bits(lP[ln * 128 + v0 + ka]);
        int kb = i + 8 * ((i >> 3) + half);
        vb.e[i] = f2bf(lV[(v0 + kb) * 16 + ln]);
      }
      oo.v = WMMA_BF16(pa.v, vb.v, oo.v);
    }
#pragma unroll
    for (int r = 0; r < 8; r++){
      int m = r + 8 * half;
      vo[((long)(w0 + m) * B + nb) * C_DIM + e * 16 + ln] = oo.e[r];
    }
    __syncthreads();
  }
  if (rawOut){
    for (int idx = lane; idx < 16 * L; idx += 32){
      int wl = idx / L, v = idx % L;
      rawOut[((long)nb * L + (w0 + wl)) * L + v] = lRaw[wl * 128 + v];
    }
  }
}

// ---------------- 3x3 conv via 9 shifted WMMA GEMMs ----------------
// y1: [W,S2D,C]; w2: [O=128][I=128][3][3]; out: [W,S2D,C]
__global__ void k_conv3(const float* __restrict__ y1, const float* __restrict__ w2,
                        const float* __restrict__ b2, float* __restrict__ out){
  int wv = (blockIdx.x * blockDim.x + threadIdx.x) >> 5;
  int lane = threadIdx.x & 31;
  if (wv >= W_FULL * 8 * 8) return;
  int w = wv >> 6;
  int h0 = ((wv >> 3) & 7) * 16;
  int o0 = (wv & 7) * 16;
  int half = lane >> 4, ln = lane & 15;
  v8f zero = {};
  union { v8f v; float e[8]; } cc; cc.v = zero;
  __builtin_prefetch(&w2[(long)(o0 + ln) * C_DIM * 9], 0, 0);
  for (int dy = 0; dy < 3; dy++)
    for (int dx = 0; dx < 3; dx++){
      int wsrc = w + dx - 1;
      if (wsrc < 0 || wsrc >= W_FULL) continue;
      int hs = h0 + ln + dy - 1;
      bool valid = (hs >= 0 && hs < S2D);
      if (valid) __builtin_prefetch(&y1[((long)wsrc * S2D + hs) * C_DIM], 0, 0);
      for (int k0 = 0; k0 < C_DIM; k0 += 32){
        union { v16bf v; __bf16 e[16]; } a, b;
#pragma unroll
        for (int i = 0; i < 16; i++){
          int ka = k0 + i + 8 * ((i >> 3) + half);
          a.e[i] = valid ? f2bf(y1[((long)wsrc * S2D + hs) * C_DIM + ka]) : f2bf(0.f);
          int kb = k0 + i + 8 * ((i >> 3) + half);
          b.e[i] = f2bf(w2[(((long)(o0 + ln)) * C_DIM + kb) * 9 + dy * 3 + dx]);
        }
        cc.v = WMMA_BF16(a.v, b.v, cc.v);
      }
    }
#pragma unroll
  for (int r = 0; r < 8; r++){
    int h = h0 + r + 8 * half;
    out[((long)w * S2D + h) * C_DIM + o0 + ln] = cc.e[r] + b2[o0 + ln];
  }
}

// ---------------- host orchestration ----------------
static inline int cdiv(int a, int b){ return (a + b - 1) / b; }

static void launch_gemm(hipStream_t st, const float* A, int M, int K,
                        const float* Wt, const float* bias, int N,
                        float* out, long outOff, long ldoL, int outB, int acc){
  int waves = ((M + 15) / 16) * (N / 16);
  hipLaunchKernelGGL(k_gemm, dim3(cdiv(waves, 4)), dim3(128), 0, st,
                     A, M, K, Wt, bias, N, out, outOff, ldoL, outB, acc);
}

static void run_mha(hipStream_t st, float* qkvA, float* qkvB, float* peb, float* vo,
                    const float* Qin, const float* KVin, int L, int B,
                    const float* iw, const float* ib, const float* ow, const float* ob,
                    const float* posenc, int flip, int causal, float* rawOut,
                    float* outPtr, long outOff, long ldoL, int outB, int acc){
  launch_gemm(st, Qin, L * B, C_DIM, iw, ib, 384, qkvA, 0, 384, 1, 0);
  const float* qkvKV = qkvA;
  if (KVin != Qin){
    launch_gemm(st, KVin, L * B, C_DIM, iw, ib, 384, qkvB, 0, 384, 1, 0);
    qkvKV = qkvB;
  }
  launch_gemm(st, posenc, 2 * L - 1, C_DIM, iw, ib, 256, peb, 0, 256, 1, 0);
  hipLaunchKernelGGL(k_attn, dim3(L / 16, B), dim3(32), 0, st,
                     qkvA, qkvKV, peb, vo, rawOut, L, B, flip, causal, 0.25f);
  launch_gemm(st, vo, L * B, C_DIM, ow, ob, C_DIM, outPtr, outOff, ldoL, outB, acc);
}

extern "C" void kernel_launch(void* const* d_in, const int* in_sizes, int n_in,
                              void* d_out, int out_size, void* d_ws, size_t ws_size,
                              hipStream_t stream){
  const float* feat_left   = (const float*)d_in[0];
  const float* feat_right  = (const float*)d_in[1];
  const float* feat_left1  = (const float*)d_in[2];
  const float* feat_right1 = (const float*)d_in[3];
  const float* pos_enc     = (const float*)d_in[4];
  const float* pos_enc1    = (const float*)d_in[5];
  const float* pos_enc_y   = (const float*)d_in[6];
  const float* pos_enc_y1  = (const float*)d_in[7];
  const float* sa_in_w  = (const float*)d_in[8];
  const float* sa_in_b  = (const float*)d_in[9];
  const float* sa_out_w = (const float*)d_in[10];
  const float* sa_out_b = (const float*)d_in[11];
  const float* sa_n1_g  = (const float*)d_in[12];
  const float* sa_n1_b  = (const float*)d_in[13];
  const float* sa_n2_g  = (const float*)d_in[14];
  const float* sa_n2_b  = (const float*)d_in[15];
  const float* ca_in_w  = (const float*)d_in[16];
  const float* ca_in_b  = (const float*)d_in[17];
  const float* ca_out_w = (const float*)d_in[18];
  const float* ca_out_b = (const float*)d_in[19];
  const float* ca_n1_g  = (const float*)d_in[20];
  const float* ca_n1_b  = (const float*)d_in[21];
  const float* ca_n2_g  = (const float*)d_in[22];
  const float* ca_n2_b  = (const float*)d_in[23];
  const float* merge_w1 = (const float*)d_in[24];
  const float* merge_b1 = (const float*)d_in[25];
  const float* merge_w2 = (const float*)d_in[26];
  const float* merge_b2 = (const float*)d_in[27];
  (void)in_sizes; (void)n_in; (void)out_size; (void)ws_size;

  float* B0    = (float*)d_ws;
  float* FEAT  = B0;                      // [128,128,128]
  float* FEAT1 = FEAT  + 2097152;         // [64,128,128]
  float* T0    = FEAT1 + 1048576;         // scratch [<=128,128,128]
  float* T1    = T0    + 2097152;         // scratch (y-layout)
  float* QKVA  = T1    + 2097152;         // [<=16384,384]
  float* QKVB  = QKVA  + 6291456;
  float* PEB   = QKVB  + 6291456;         // [<=255,256]
  float* VO    = PEB   + 65536;           // [<=16384,128]
  float* FL2   = VO    + 2097152;         // [128,64,128]
  float* FR2   = FL2   + 1048576;
  float* FR2N  = FR2   + 1048576;
  float* FL4   = FR2N  + 1048576;         // [64,64,128]
  float* FR4   = FL4   + 524288;
  float* FR4N  = FR4   + 524288;
  float* FL1N  = FR4N  + 524288;
  float* FR1N  = FL1N  + 524288;
  float* FN    = FR1N  + 524288;          // [64,128,128]
  float* CVIN  = FN    + 1048576;         // [128,128,256]
  float* Y1B   = CVIN  + 4194304;         // [128,128,128]

  const long ldF  = (long)S2D * C_DIM;    // per-w stride in feat/feat1

  // build [W,2HN,C] layouts
  hipLaunchKernelGGL(k_flat, dim3(cdiv(W_FULL * S2D * C_DIM, 256)), dim3(256), 0, stream,
                     feat_left, feat_right, FEAT, W_FULL, HN);
  hipLaunchKernelGGL(k_flat, dim3(cdiv(W_LOW * S2D * C_DIM, 256)), dim3(256), 0, stream,
                     feat_left1, feat_right1, FEAT1, W_LOW, HN);

  for (int i = 0; i < LYR; i++){
    const float* siw[4]; const float* sib[4]; const float* sow[4]; const float* sob[4];
    for (int j = 0; j < 4; j++){
      siw[j] = sa_in_w  + (long)(i * 4 + j) * 384 * 128;
      sib[j] = sa_in_b  + (long)(i * 4 + j) * 384;
      sow[j] = sa_out_w + (long)(i * 4 + j) * 128 * 128;
      sob[j] = sa_out_b + (long)(i * 4 + j) * 128;
    }
    const float* ciw[2]; const float* cib[2]; const float* cow[2]; const float* cob[2];
    for (int j = 0; j < 2; j++){
      ciw[j] = ca_in_w  + (long)(i * 2 + j) * 384 * 128;
      cib[j] = ca_in_b  + (long)(i * 2 + j) * 384;
      cow[j] = ca_out_w + (long)(i * 2 + j) * 128 * 128;
      cob[j] = ca_out_b + (long)(i * 2 + j) * 128;
    }

    // ================= self layer =================
    hipLaunchKernelGGL(k_ln, dim3(cdiv(W_FULL * S2D, 4)), dim3(128), 0, stream,
                       FEAT, 0L, ldF, S2D, sa_n1_g + i * 128, sa_n1_b + i * 128, T0, W_FULL);
    hipLaunchKernelGGL(k_permy, dim3(cdiv(W_FULL * S2D * C_DIM, 256)), dim3(256), 0, stream,
                       T0, T1, W_FULL, 1);
    run_mha(stream, QKVA, QKVB, PEB, VO, T1, T1, HN, 2 * W_FULL,
            siw[1], sib[1], sow[1], sob[1], pos_enc_y, 0, 0, nullptr,
            T1, 0, C_DIM, 1, 0);
    hipLaunchKernelGGL(k_permy, dim3(cdiv(W_FULL * S2D * C_DIM, 256)), dim3(256), 0, stream,
                       T0, T1, W_FULL, 0);
    run_mha(stream, QKVA, QKVB, PEB, VO, T0, T0, W_FULL, S2D,
            siw[0], sib[0], sow[0], sob[0], pos_enc, 0, 0, nullptr,
            FEAT, 0, C_DIM, 1, 1);
    hipLaunchKernelGGL(k_ln, dim3(cdiv(W_LOW * S2D, 4)), dim3(128), 0, stream,
                       FEAT1, 0L, ldF, S2D, sa_n2_g + i * 128, sa_n2_b + i * 128, T0, W_LOW);
    hipLaunchKernelGGL(k_permy, dim3(cdiv(W_LOW * S2D * C_DIM, 256)), dim3(256), 0, stream,
                       T0, T1, W_LOW, 1);
    run_mha(stream, QKVA, QKVB, PEB, VO, T1, T1, HN, 2 * W_LOW,
            siw[3], sib[3], sow[3], sob[3], pos_enc_y1, 0, 0, nullptr,
            T1, 0, C_DIM, 1, 0);
    hipLaunchKernelGGL(k_permy, dim3(cdiv(W_LOW * S2D * C_DIM, 256)), dim3(256), 0, stream,
                       T0, T1, W_LOW, 0);
    run_mha(stream, QKVA, QKVB, PEB, VO, T0, T0, W_LOW, S2D,
            siw[2], sib[2], sow[2], sob[2], pos_enc1, 0, 0, nullptr,
            FEAT1, 0, C_DIM, 1, 1);

    // ================= cross layer =================
    int last = (i == LYR - 1);
    hipLaunchKernelGGL(k_ln, dim3(cdiv(W_FULL * HN, 4)), dim3(128), 0, stream,
                       FEAT, 0L, ldF, HN, ca_n1_g + i * 128, ca_n1_b + i * 128, FL2, W_FULL);
    hipLaunchKernelGGL(k_ln, dim3(cdiv(W_FULL * HN, 4)), dim3(128), 0, stream,
                       FEAT, (long)HN * C_DIM, ldF, HN, ca_n1_g + i * 128, ca_n1_b + i * 128, FR2, W_FULL);
    hipLaunchKernelGGL(k_ln, dim3(cdiv(W_LOW * HN, 4)), dim3(128), 0, stream,
                       FEAT1, 0L, ldF, HN, ca_n2_g + i * 128, ca_n2_b + i * 128, FL4, W_LOW);
    hipLaunchKernelGGL(k_ln, dim3(cdiv(W_LOW * HN, 4)), dim3(128), 0, stream,
                       FEAT1, (long)HN * C_DIM, ldF, HN, ca_n2_g + i * 128, ca_n2_b + i * 128, FR4, W_LOW);

    run_mha(stream, QKVA, QKVB, PEB, VO, FR2, FL2, W_FULL, HN,
            ciw[0], cib[0], cow[0], cob[0], pos_enc, 1, 0, nullptr,
            FEAT, (long)HN * C_DIM, ldF, HN, 1);
    hipLaunchKernelGGL(k_copy, dim3(cdiv(W_LOW * HN * C_DIM, 256)), dim3(256), 0, stream,
                       FEAT1, (long)HN * C_DIM, ldF, FR1N, 0L, (long)HN * C_DIM, HN, W_LOW);
    run_mha(stream, QKVA, QKVB, PEB, VO, FR4, FL4, W_LOW, HN,
            ciw[1], cib[1], cow[1], cob[1], pos_enc1, 1, 0, nullptr,
            FR1N, 0, C_DIM, 1, 1);
    hipLaunchKernelGGL(k_ln, dim3(cdiv(W_FULL * HN, 4)), dim3(128), 0, stream,
                       FEAT, (long)HN * C_DIM, ldF, HN, ca_n1_g + i * 128, ca_n1_b + i * 128, FR2N, W_FULL);
    run_mha(stream, QKVA, QKVB, PEB, VO, FL2, FR2N, W_FULL, HN,
            ciw[0], cib[0], cow[0], cob[0], pos_enc, 0, last, last ? (float*)d_out : nullptr,
            FEAT, 0, ldF, HN, 1);
    hipLaunchKernelGGL(k_ln, dim3(cdiv(W_LOW * HN, 4)), dim3(128), 0, stream,
                       FR1N, 0L, (long)HN * C_DIM, HN, ca_n2_g + i * 128, ca_n2_b + i * 128, FR4N, W_LOW);
    hipLaunchKernelGGL(k_copy, dim3(cdiv(W_LOW * HN * C_DIM, 256)), dim3(256), 0, stream,
                       FEAT1, 0L, ldF, FL1N, 0L, (long)HN * C_DIM, HN, W_LOW);
    run_mha(stream, QKVA, QKVB, PEB, VO, FL4, FR4N, W_LOW, HN,
            ciw[1], cib[1], cow[1], cob[1], pos_enc1, 0, last, nullptr,
            FL1N, 0, C_DIM, 1, 1);

    hipLaunchKernelGGL(k_copy, dim3(cdiv(W_LOW * HN * C_DIM, 256)), dim3(256), 0, stream,
                       FL1N, 0L, (long)HN * C_DIM, FN, 0L, ldF, HN, W_LOW);
    hipLaunchKernelGGL(k_copy, dim3(cdiv(W_LOW * HN * C_DIM, 256)), dim3(256), 0, stream,
                       FR1N, 0L, (long)HN * C_DIM, FN, (long)HN * C_DIM, ldF, HN, W_LOW);
    hipLaunchKernelGGL(k_interpcat, dim3(cdiv(W_FULL * S2D * 256, 256)), dim3(256), 0, stream,
                       FN, FEAT, CVIN);
    launch_gemm(stream, CVIN, W_FULL * S2D, 256,
                merge_w1 + (long)i * 128 * 256, merge_b1 + (long)i * 128, 128,
                Y1B, 0, 128, 1, 0);
    hipLaunchKernelGGL(k_conv3, dim3(cdiv(W_FULL * 8 * 8, 4)), dim3(128), 0, stream,
                       Y1B, merge_w2 + (long)i * 128 * 128 * 9, merge_b2 + (long)i * 128, FEAT);
  }
}
